// NestedGIN_73985106641258
// MI455X (gfx1250) — matrix operand order repaired and checked by reading, not compiled, not run on hardware
//
#include <hip/hip_runtime.h>
#include <hip/hip_bf16.h>
#include <math.h>

// ---------------------------------------------------------------------------
// NestedGIN forward for MI455X (gfx1250, wave32, WMMA f16->f32, TDM weights)
// ---------------------------------------------------------------------------

typedef __attribute__((ext_vector_type(16))) _Float16 v16h;
typedef __attribute__((ext_vector_type(8)))  float    v8f;
typedef __attribute__((ext_vector_type(4)))  unsigned int v4u;
typedef __attribute__((ext_vector_type(8)))  int      v8i;
typedef __attribute__((ext_vector_type(4)))  int      v4i;

#define HDIM 128
#define WSTR 136                      // padded halves per row of W^T
#define WPANEL (HDIM * WSTR)          // 17408 halves = 34816 bytes per weight
#define RSQ  0.9999950000374996f      // 1/sqrt(1 + 1e-5)  (eval BatchNorm)

static constexpr int   cN = 50000;
static constexpr int   cE = 800000;
static constexpr long  cP = 3200000;
static constexpr int   cG = 100;
static constexpr int   cL = 3;

#ifndef __has_builtin
#define __has_builtin(x) 0
#endif
#if __has_builtin(__builtin_amdgcn_tensor_load_to_lds) && \
    __has_builtin(__builtin_amdgcn_s_wait_tensorcnt)
#define USE_TDM 1
#else
#define USE_TDM 0
#endif

#if USE_TDM
// DMA one contiguous 34816B f16 W^T panel from global into LDS via the
// Tensor Data Mover.  1-D tile, data_size = 8 bytes -> 4352 units.
__device__ __forceinline__ void tdm_load_panel(const _Float16* gW,
                                               _Float16* sW) {
  unsigned long long ga = (unsigned long long)(size_t)gW;
  unsigned int lds = (unsigned int)(size_t)sW;      // addrspace(3) byte offset
  // D# group 0: count=1 | lds_addr | global_addr[56:0] | type=2
  v4u g0 = { 1u, lds, (unsigned int)ga,
             (((unsigned int)(ga >> 32)) & 0x01FFFFFFu) | 0x80000000u };
  // D# group 1: data_size=8B; tensor_dim0 = tile_dim0 = 4352; dim1 = 1
  v8i g1 = { 0x00030000, 0x11000000, 0x00010000, 0x11000000,
             1, 4352, 0, 0 };
  v4i g2 = {};
  v4i g3 = {};
#if __clang_major__ >= 23
  v8i g4 = {};
  __builtin_amdgcn_tensor_load_to_lds(g0, g1, g2, g3, g4, 0);
#else
  __builtin_amdgcn_tensor_load_to_lds(g0, g1, g2, g3, 0);
#endif
  __builtin_amdgcn_s_wait_tensorcnt(0);
}
#endif

// ---------------------------------------------------------------- utilities
__global__ void k_fill(float* p, float v, long n) {
  long i = (long)blockIdx.x * blockDim.x + threadIdx.x;
  if (i < n) p[i] = v;
}

// Pre-transpose + convert one weight matrix: W[k][n] f32 -> Wt[n*WSTR+k] f16,
// pad columns 128..135 with zeros (TDM copies the whole panel).
__global__ void k_prep_w(const float* W, _Float16* out) {
  int i = blockIdx.x * blockDim.x + threadIdx.x;
  if (i >= WPANEL) return;
  int n = i / WSTR;
  int k = i - n * WSTR;
  out[i] = (k < HDIM) ? (_Float16)W[k * HDIM + n] : (_Float16)0.f;
}

// z[pos_batch[p], h] += z_table[pos_index[p], h] * pos_enc[p]
__global__ void k_scatter_pos(const int* pos_index, const float* pos_enc,
                              const int* pos_batch, const float* z_table,
                              float* z, long total) {
  long i = (long)blockIdx.x * blockDim.x + threadIdx.x;
  if (i >= total) return;
  long p = i >> 7;
  int  h = (int)(i & 127);
  float v = z_table[(size_t)pos_index[p] * HDIM + h] * pos_enc[p];
  atomicAdd(&z[(size_t)pos_batch[p] * HDIM + h], v);
}

// ----------------------------------------------------- generic WMMA GEMM 128
// AMODE: 1 = f32 A with relu(A*s_in + b_in) (BN+ReLU prologue),
//        2 = f32 A plus second f32 addend A2, 3 = f16 A.
// EPI:   0 = relu(acc + bias) -> f32, 1 = relu((acc+bias)*s_out + b_out) -> f16,
//        2 = relu(acc + bias) -> f16.
template <int AMODE, int EPI>
__global__ __launch_bounds__(256)
void k_gemm(const void* A, const float* A2, const _Float16* Wt,
            const float* bias, const float* g_in, const float* b_in,
            const float* g_out, const float* b_out,
            void* Out, int nrows) {
  __shared__ alignas(16) _Float16 sW[WPANEL];
  __shared__ float sSc[HDIM];
  __shared__ float sBi[HDIM];

  const int tid  = threadIdx.x;
  const int lane = tid & 31;
  const int wave = tid >> 5;                       // 0..7
  const int rowBase = blockIdx.x * 128 + wave * 16;

#if USE_TDM
  if (wave == 0 && lane == 0) tdm_load_panel(Wt, sW);
#else
  for (int i = tid; i < WPANEL; i += 256) sW[i] = Wt[i];
#endif
  if (AMODE == 1 && tid < HDIM) {
    sSc[tid] = g_in[tid] * RSQ;
    sBi[tid] = b_in[tid];
  }
  __syncthreads();

  const int half = lane >> 4;          // K-phase select for A / B
  const int m    = lane & 15;          // A row within tile, also C/D column
  int arow  = rowBase + m;
  int arowc = arow < nrows ? arow : nrows - 1;     // clamp, keep EXEC uniform

  // ---- preload full K=128 strip of A (16 rows per wave) ----
  v16h a[4];
#pragma unroll
  for (int k4 = 0; k4 < 4; ++k4) {
    int kb0 = k4 * 32 + half * 8;      // halves 0..7  -> K = kb0 + i
    int kb1 = kb0 + 16;                // halves 8..15 -> K = kb1 + i
    if (AMODE == 3) {
      const _Float16* Ah = (const _Float16*)A + (size_t)arowc * HDIM;
#pragma unroll
      for (int i = 0; i < 8; ++i) {
        a[k4][i]     = Ah[kb0 + i];
        a[k4][8 + i] = Ah[kb1 + i];
      }
    } else {
      const float* Af = (const float*)A + (size_t)arowc * HDIM;
#pragma unroll
      for (int i = 0; i < 8; ++i) {
        float v0 = Af[kb0 + i];
        float v1 = Af[kb1 + i];
        if (AMODE == 1) {
          v0 = fmaxf(v0 * sSc[kb0 + i] + sBi[kb0 + i], 0.f);
          v1 = fmaxf(v1 * sSc[kb1 + i] + sBi[kb1 + i], 0.f);
        }
        if (AMODE == 2) {
          v0 += A2[(size_t)arowc * HDIM + kb0 + i];
          v1 += A2[(size_t)arowc * HDIM + kb1 + i];
        }
        a[k4][i]     = (_Float16)v0;
        a[k4][8 + i] = (_Float16)v1;
      }
    }
  }

  // ---- 8 column tiles x 4 K-steps ----
#pragma unroll
  for (int n0 = 0; n0 < 8; ++n0) {
    v8f acc = {};
#pragma unroll
    for (int k4 = 0; k4 < 4; ++k4) {
      v16h b;
      const _Float16* bp = sW + (size_t)(n0 * 16 + m) * WSTR + k4 * 32 + half * 8;
#pragma unroll
      for (int i = 0; i < 8; ++i) { b[i] = bp[i]; b[8 + i] = bp[16 + i]; }
      acc = __builtin_amdgcn_wmma_f32_16x16x32_f16(
          false, a[k4], false, b, (short)0, acc, false, false);
    }
    int col = n0 * 16 + m;
    float bv = bias[col];
    float so = 1.f, bo = 0.f;
    if (EPI == 1) { so = g_out[col] * RSQ; bo = b_out[col]; }
#pragma unroll
    for (int r = 0; r < 8; ++r) {
      int row = rowBase + half * 8 + r;
      if (row < nrows) {
        float y = acc[r] + bv;
        if (EPI == 1) y = y * so + bo;
        y = fmaxf(y, 0.f);
        if (EPI == 0) ((float*)Out)[(size_t)row * HDIM + col] = y;
        else          ((_Float16*)Out)[(size_t)row * HDIM + col] = (_Float16)y;
      }
    }
  }
}

// ------------------------------------------ edge GEMM + fused message scatter
// e = z2 @ We + be ; msg = relu(x[src] + e) ; hacc[dst] += msg   (atomic)
__global__ __launch_bounds__(256)
void k_gemm_edge(const _Float16* z2, const _Float16* Wt, const float* be,
                 const int* src, const int* dst, const float* x, float* hacc) {
  __shared__ alignas(16) _Float16 sW[WPANEL];
  const int tid  = threadIdx.x;
  const int lane = tid & 31;
  const int wave = tid >> 5;
  const int rowBase = blockIdx.x * 128 + wave * 16;   // E % 128 == 0

#if USE_TDM
  if (wave == 0 && lane == 0) tdm_load_panel(Wt, sW);
#else
  for (int i = tid; i < WPANEL; i += 256) sW[i] = Wt[i];
#endif
  __syncthreads();

  const int half = lane >> 4;
  const int m    = lane & 15;
  const size_t arow = (size_t)(rowBase + m);

  v16h a[4];
#pragma unroll
  for (int k4 = 0; k4 < 4; ++k4) {
    int kb0 = k4 * 32 + half * 8;
    int kb1 = kb0 + 16;
    const _Float16* Ah = z2 + arow * HDIM;
#pragma unroll
    for (int i = 0; i < 8; ++i) {
      a[k4][i]     = Ah[kb0 + i];
      a[k4][8 + i] = Ah[kb1 + i];
    }
  }

  // edge endpoints for the 8 C-rows this half-wave owns
  int se[8], de[8];
#pragma unroll
  for (int r = 0; r < 8; ++r) {
    int e = rowBase + half * 8 + r;
    se[r] = src[e];
    de[r] = dst[e];
  }

#pragma unroll
  for (int n0 = 0; n0 < 8; ++n0) {
    v8f acc = {};
#pragma unroll
    for (int k4 = 0; k4 < 4; ++k4) {
      v16h b;
      const _Float16* bp = sW + (size_t)(n0 * 16 + m) * WSTR + k4 * 32 + half * 8;
#pragma unroll
      for (int i = 0; i < 8; ++i) { b[i] = bp[i]; b[8 + i] = bp[16 + i]; }
      acc = __builtin_amdgcn_wmma_f32_16x16x32_f16(
          false, a[k4], false, b, (short)0, acc, false, false);
    }
    int col = n0 * 16 + m;
    float bv = be[col];
#pragma unroll
    for (int r = 0; r < 8; ++r) {
      float msg = fmaxf(x[(size_t)se[r] * HDIM + col] + acc[r] + bv, 0.f);
      atomicAdd(&hacc[(size_t)de[r] * HDIM + col], msg);
    }
  }
}

// ------------------------------------------------- conv1 (H->1 edge channel)
// one wave per edge: e1 = dot(z2[e], We1) ; hacc1[dst] += relu(1 + e1 + be1)
__global__ __launch_bounds__(256)
void k_conv1_edge(const _Float16* z2, const float* We1, const float* be1,
                  const int* src, const int* dst, float* hacc1) {
  int lane = threadIdx.x & 31;
  int e = blockIdx.x * 8 + (threadIdx.x >> 5);
  const _Float16* row = z2 + (size_t)e * HDIM;
  float s = 0.f;
#pragma unroll
  for (int j = 0; j < 4; ++j) {
    int k = lane + 32 * j;
    s += (float)row[k] * We1[k];
  }
#pragma unroll
  for (int off = 16; off > 0; off >>= 1) s += __shfl_xor(s, off, 32);
  if (lane == 0) {
    float msg = fmaxf(1.0f + s + be1[0], 0.f);   // x[src] == 1
    atomicAdd(&hacc1[dst[e]], msg);
  }
}

// x1[n,h] = relu(h1[n] * W1a[h] + b1a[h])  stored f16
__global__ void k_conv1_expand(const float* hacc1, const float* W1a,
                               const float* b1a, _Float16* x1, long total) {
  long i = (long)blockIdx.x * blockDim.x + threadIdx.x;
  if (i >= total) return;
  long n = i >> 7;
  int  h = (int)(i & 127);
  x1[i] = (_Float16)fmaxf(hacc1[n] * W1a[h] + b1a[h], 0.f);
}

// ---------------------------------------------------------------- pooling
__global__ void k_pool(const float* x, const int* batch, float* g, long total) {
  long i = (long)blockIdx.x * blockDim.x + threadIdx.x;
  if (i >= total) return;
  long n = i >> 7;
  int  h = (int)(i & 127);
  atomicAdd(&g[(size_t)batch[n] * HDIM + h], x[i]);
}

// ------------------------------------- readout heads + log_softmax (1 blk/graph)
__global__ __launch_bounds__(128)
void k_readout(const float* g, const float* Wl1, const float* bl1,
               const float* Wl2, const float* bl2, float* out) {
  __shared__ float sg[HDIM];
  __shared__ float sh1[HDIM];
  __shared__ float red[HDIM];
  int t = threadIdx.x;
  int gr = blockIdx.x;
  sg[t] = g[(size_t)gr * HDIM + t];
  __syncthreads();
  float a1 = bl1[t];
  for (int k = 0; k < HDIM; ++k) a1 += sg[k] * Wl1[k * HDIM + t];
  sh1[t] = fmaxf(a1, 0.f);
  __syncthreads();
  float lg = bl2[t];
  for (int k = 0; k < HDIM; ++k) lg += sh1[k] * Wl2[k * HDIM + t];
  red[t] = lg;
  __syncthreads();
  for (int s = 64; s > 0; s >>= 1) {
    if (t < s) red[t] = fmaxf(red[t], red[t + s]);
    __syncthreads();
  }
  float mx = red[0];
  __syncthreads();
  red[t] = __expf(lg - mx);
  __syncthreads();
  for (int s = 64; s > 0; s >>= 1) {
    if (t < s) red[t] += red[t + s];
    __syncthreads();
  }
  float lse = mx + __logf(red[0]);
  out[(size_t)gr * HDIM + t] = lg - lse;
}

// ---------------------------------------------------------------------------
extern "C" void kernel_launch(void* const* d_in, const int* in_sizes, int n_in,
                              void* d_out, int out_size, void* d_ws, size_t ws_size,
                              hipStream_t stream) {
  (void)in_sizes; (void)n_in; (void)out_size; (void)ws_size;

  const int*   edge_index = (const int*)  d_in[0];
  const int*   batch      = (const int*)  d_in[1];
  const int*   pos_index  = (const int*)  d_in[2];
  const float* pos_enc    = (const float*)d_in[3];
  const int*   pos_batch  = (const int*)  d_in[4];
  const float* z_table    = (const float*)d_in[5];
  const float* bn1_g      = (const float*)d_in[6];
  const float* bn1_b      = (const float*)d_in[7];
  const float* Wz         = (const float*)d_in[8];
  const float* bz         = (const float*)d_in[9];
  const float* bn2_g      = (const float*)d_in[10];
  const float* bn2_b      = (const float*)d_in[11];
  const float* We1        = (const float*)d_in[12];
  const float* be1        = (const float*)d_in[13];
  const float* W1a        = (const float*)d_in[14];
  const float* b1a        = (const float*)d_in[15];
  const float* W1b        = (const float*)d_in[16];
  const float* b1b        = (const float*)d_in[17];
  const float* We         = (const float*)d_in[18];
  const float* be         = (const float*)d_in[19];
  const float* Wa         = (const float*)d_in[20];
  const float* ba         = (const float*)d_in[21];
  const float* Wb         = (const float*)d_in[22];
  const float* bb         = (const float*)d_in[23];
  const float* Wl1        = (const float*)d_in[24];
  const float* bl1        = (const float*)d_in[25];
  const float* Wl2        = (const float*)d_in[26];
  const float* bl2        = (const float*)d_in[27];

  const int* src = edge_index;          // edge_index[0,:]
  const int* dst = edge_index + cE;     // edge_index[1,:]

  // -------- workspace carve-up --------
  auto up = [](size_t x) { return (x + 255) & ~(size_t)255; };
  char* ws = (char*)d_ws;
  size_t off = 0;
  float*    z     = (float*)(ws + off);    off += up((size_t)cE * HDIM * 4);
  _Float16* z2    = (_Float16*)(ws + off); off += up((size_t)cE * HDIM * 2);
  float*    hacc1 = (float*)(ws + off);    off += up((size_t)cN * 4);
  float*    hacc  = (float*)(ws + off);    off += up((size_t)cN * HDIM * 4);
  float*    x     = (float*)(ws + off);    off += up((size_t)cN * HDIM * 4);
  _Float16* tmp16 = (_Float16*)(ws + off); off += up((size_t)cN * HDIM * 2);
  float*    gpool = (float*)(ws + off);    off += up((size_t)cG * HDIM * 4);
  // 11 pre-transposed f16 weight panels: Wz, W1b, We[0..2], Wa[0..2], Wb[0..2]
  _Float16* wt    = (_Float16*)(ws + off); off += up((size_t)11 * WPANEL * 2);
  _Float16* wtWz  = wt + 0 * WPANEL;
  _Float16* wtW1b = wt + 1 * WPANEL;
  _Float16* wtWe  = wt + 2 * WPANEL;     // 3 panels
  _Float16* wtWa  = wt + 5 * WPANEL;     // 3 panels
  _Float16* wtWb  = wt + 8 * WPANEL;     // 3 panels

  const long zElems = (long)cE * HDIM;
  const long nElems = (long)cN * HDIM;
  const int  gridE  = cE / 128;                  // 6250
  const int  gridN  = (cN + 127) / 128;          // 391
  const int  gridW  = (WPANEL + 255) / 256;      // 68

  // 0) prep f16 transposed weight panels (DMA'd into LDS by GEMM blocks)
  k_prep_w<<<gridW, 256, 0, stream>>>(Wz,  wtWz);
  k_prep_w<<<gridW, 256, 0, stream>>>(W1b, wtW1b);
  for (int l = 0; l < cL; ++l) {
    k_prep_w<<<gridW, 256, 0, stream>>>(We + (size_t)l * HDIM * HDIM, wtWe + (size_t)l * WPANEL);
    k_prep_w<<<gridW, 256, 0, stream>>>(Wa + (size_t)l * HDIM * HDIM, wtWa + (size_t)l * WPANEL);
    k_prep_w<<<gridW, 256, 0, stream>>>(Wb + (size_t)l * HDIM * HDIM, wtWb + (size_t)l * WPANEL);
  }

  // 1) z = segment_sum(z_table[pos_index] * pos_enc, pos_batch)
  k_fill<<<(int)((zElems + 255) / 256), 256, 0, stream>>>(z, 0.f, zElems);
  long pTot = cP * (long)HDIM;
  k_scatter_pos<<<(int)((pTot + 255) / 256), 256, 0, stream>>>(
      pos_index, pos_enc, pos_batch, z_table, z, pTot);

  // 2) z2 = relu(bn2(relu(bn1(z)) @ Wz + bz))   [E,128] f16
  k_gemm<1, 1><<<gridE, 256, 0, stream>>>(
      z, nullptr, wtWz, bz, bn1_g, bn1_b, bn2_g, bn2_b, z2, cE);

  // 3) conv1: h1 = 1 + segsum(relu(1 + z2@We1 + be1)); x = MLP(h1)
  k_fill<<<(int)((cN + 255) / 256), 256, 0, stream>>>(hacc1, 1.f, cN);
  k_conv1_edge<<<cE / 8, 256, 0, stream>>>(z2, We1, be1, src, dst, hacc1);
  k_conv1_expand<<<(int)((nElems + 255) / 256), 256, 0, stream>>>(
      hacc1, W1a, b1a, tmp16, nElems);
  k_gemm<3, 0><<<gridN, 256, 0, stream>>>(
      tmp16, nullptr, wtW1b, b1b, nullptr, nullptr, nullptr, nullptr, x, cN);

  // 4) three GINEConv layers
  for (int l = 0; l < cL; ++l) {
    const float* bel = be + (size_t)l * HDIM;
    const float* bal = ba + (size_t)l * HDIM;
    const float* bbl = bb + (size_t)l * HDIM;

    k_fill<<<(int)((nElems + 255) / 256), 256, 0, stream>>>(hacc, 0.f, nElems);
    k_gemm_edge<<<gridE, 256, 0, stream>>>(
        z2, wtWe + (size_t)l * WPANEL, bel, src, dst, x, hacc);
    // tmp16 = relu((x + hacc) @ Wa + ba)   [N,128] f16
    k_gemm<2, 2><<<gridN, 256, 0, stream>>>(
        x, hacc, wtWa + (size_t)l * WPANEL, bal,
        nullptr, nullptr, nullptr, nullptr, tmp16, cN);
    // x = relu(tmp16 @ Wb + bb)            [N,128] f32
    k_gemm<3, 0><<<gridN, 256, 0, stream>>>(
        tmp16, nullptr, wtWb + (size_t)l * WPANEL, bbl,
        nullptr, nullptr, nullptr, nullptr, x, cN);
  }

  // 5) readout
  k_fill<<<(cG * HDIM + 255) / 256, 256, 0, stream>>>(gpool, 0.f, (long)cG * HDIM);
  k_pool<<<(int)((nElems + 255) / 256), 256, 0, stream>>>(x, batch, gpool, nElems);
  k_readout<<<cG, 128, 0, stream>>>(gpool, Wl1, bl1, Wl2, bl2, (float*)d_out);
}
